// DAGR_60773787238415
// MI455X (gfx1250) — compile-verified, matrix operand order checked by reading, not run on hardware
//
#include <hip/hip_runtime.h>
#include <hip/hip_bf16.h>

#define B_IMG 32
#define N_BOX 2048
#define C_CH  85
#define NCLS  80
#define CONF_THRES 0.25f
#define NMS_THRES  0.45f
#define MIN_SIZE   5.0f

typedef __attribute__((ext_vector_type(16))) _Float16 v16h;
typedef __attribute__((ext_vector_type(8)))  float    v8f;

// ---------------------------------------------------------------------------
// K1: per-image statistics: any(conf>=thres), top-5 fallback indices, valid[],
//     xywh vs xyxy vote -> use_xywh[b]
// ---------------------------------------------------------------------------
__global__ void k1_stats(const float* __restrict__ pred,
                         int* __restrict__ validA,
                         int* __restrict__ useXY) {
  const int b = blockIdx.x;
  const int t = threadIdx.x;                     // 256 threads
  __shared__ float rv[256];
  __shared__ int   ri[256];
  __shared__ int   topIdx[5];
  __shared__ int   s_any;
  __shared__ int   s_xywh;

  const float* p = pred + (size_t)b * N_BOX * C_CH;

  // count boxes above threshold
  int cnt = 0;
  for (int n = t; n < N_BOX; n += 256)
    cnt += (p[(size_t)n * C_CH + 4] >= CONF_THRES) ? 1 : 0;
  ri[t] = cnt; __syncthreads();
  for (int s = 128; s > 0; s >>= 1) { if (t < s) ri[t] += ri[t + s]; __syncthreads(); }
  if (t == 0) s_any = (ri[0] > 0) ? 1 : 0;
  __syncthreads();
  const int anyAbove = s_any;

  // top-5 conf indices (5 masked argmax reductions; tie -> lower index)
  for (int k = 0; k < 5; k++) {
    float bv = -1e30f; int bi = N_BOX;
    for (int n = t; n < N_BOX; n += 256) {
      bool chosen = false;
      for (int q = 0; q < k; q++) chosen |= (topIdx[q] == n);
      float v = p[(size_t)n * C_CH + 4];
      if (!chosen && (v > bv || (v == bv && n < bi))) { bv = v; bi = n; }
    }
    rv[t] = bv; ri[t] = bi; __syncthreads();
    for (int s = 128; s > 0; s >>= 1) {
      if (t < s) {
        if (rv[t + s] > rv[t] || (rv[t + s] == rv[t] && ri[t + s] < ri[t])) {
          rv[t] = rv[t + s]; ri[t] = ri[t + s];
        }
      }
      __syncthreads();
    }
    if (t == 0) topIdx[k] = ri[0];
    __syncthreads();
  }

  // valid[] and the xywh/xyxy vote
  int xywhS = 0, xyxyS = 0;
  for (int n = t; n < N_BOX; n += 256) {
    const float* row = p + (size_t)n * C_CH;
    float conf = row[4];
    bool top = false;
    for (int q = 0; q < 5; q++) top |= (topIdx[q] == n);
    int v = anyAbove ? (conf >= CONF_THRES ? 1 : 0) : (top ? 1 : 0);
    validA[(size_t)b * N_BOX + n] = v;
    float cx = row[0], cy = row[1], pw = row[2], ph = row[3];
    if (v) {
      xywhS += ((pw > 0.f) && (ph > 0.f)) ? 1 : 0;
      xyxyS += ((pw > cx) && (ph > cy)) ? 1 : 0;
    }
  }
  ri[t] = xywhS; __syncthreads();
  for (int s = 128; s > 0; s >>= 1) { if (t < s) ri[t] += ri[t + s]; __syncthreads(); }
  if (t == 0) s_xywh = ri[0];
  __syncthreads();
  ri[t] = xyxyS; __syncthreads();
  for (int s = 128; s > 0; s >>= 1) { if (t < s) ri[t] += ri[t + s]; __syncthreads(); }
  if (t == 0) useXY[b] = (s_xywh >= ri[0]) ? 1 : 0;
}

// ---------------------------------------------------------------------------
// K2: per-box decode, class max/argmax, sort keys, out cols 0..5 and labels
// ---------------------------------------------------------------------------
__global__ void k2_decode(const float* __restrict__ pred,
                          const int* __restrict__ validA,
                          const int* __restrict__ useXY,
                          float4* __restrict__ boxes,
                          int* __restrict__ fvalid,
                          int* __restrict__ labels,
                          unsigned* __restrict__ skey,
                          float* __restrict__ out,
                          float* __restrict__ outLab) {
  int idx = blockIdx.x * 256 + threadIdx.x;
  if (idx >= B_IMG * N_BOX) return;
  int b = idx / N_BOX;
  const float* p = pred + (size_t)idx * C_CH;
  float cx = p[0], cy = p[1], pw = p[2], ph = p[3], conf = p[4];

  float mc = p[5]; int mi = 0;
  for (int c = 1; c < NCLS; c++) { float v = p[5 + c]; if (v > mc) { mc = v; mi = c; } }

  int use = useXY[b];
  int v   = validA[idx];

  float w = fmaxf(pw, MIN_SIZE), h = fmaxf(ph, MIN_SIZE);
  float bx1 = cx - 0.5f * w, by1 = cy - 0.5f * h;
  float bx2 = cx + 0.5f * w, by2 = cy + 0.5f * h;
  float area = (bx2 - bx1) * (by2 - by1);
  bool poswh = (pw > 0.f) && (ph > 0.f);
  bool reas  = (pw < 2000.f) && (ph < 2000.f);
  bool xywhValid = poswh && reas && (area >= MIN_SIZE * MIN_SIZE);
  bool vxyxy = (pw > cx) && (ph > cy);

  float x1, y1, x2, y2; int bval;
  if (use) { x1 = bx1; y1 = by1; x2 = bx2; y2 = by2; bval = xywhValid ? 1 : 0; }
  else     { x1 = cx;  y1 = cy;  x2 = pw;  y2 = ph;  bval = vxyxy ? 1 : 0; }
  int fv = (v && bval) ? 1 : 0;

  boxes[idx]  = make_float4(x1, y1, x2, y2);
  fvalid[idx] = fv;
  labels[idx] = mi;

  // sort key: ascending on ~ordered(score) with score = fv ? conf : -inf,
  // tie-broken by ascending index  ==  stable argsort of descending score
  float skf = fv ? conf : -__builtin_inff();
  unsigned bits = __float_as_uint(skf);
  unsigned ou = (bits & 0x80000000u) ? ~bits : (bits | 0x80000000u);
  skey[idx] = ~ou;

  float* o = out + (size_t)idx * 7;
  o[0] = x1; o[1] = y1; o[2] = x2; o[3] = y2; o[4] = conf; o[5] = mc;
  outLab[idx] = (float)mi;
}

// ---------------------------------------------------------------------------
// K3: per-image bitonic argsort of 64-bit keys in LDS (2048 elems, 1024 thr)
// ---------------------------------------------------------------------------
__global__ void __launch_bounds__(1024) k3_sort(const unsigned* __restrict__ skey,
                                                int* __restrict__ order) {
  const int b = blockIdx.x, t = threadIdx.x;
  __shared__ unsigned long long keys[N_BOX];
  for (int e = 0; e < 2; e++) {
    int i = t + e * 1024;
    keys[i] = ((unsigned long long)skey[(size_t)b * N_BOX + i] << 32) | (unsigned)i;
  }
  __syncthreads();
  for (int k = 2; k <= N_BOX; k <<= 1) {
    for (int j = k >> 1; j > 0; j >>= 1) {
      for (int e = 0; e < 2; e++) {
        int i = t + e * 1024;
        int ixj = i ^ j;
        if (ixj > i) {
          bool up = ((i & k) == 0);
          unsigned long long a = keys[i], c = keys[ixj];
          bool sw = up ? (a > c) : (a < c);
          if (sw) { keys[i] = c; keys[ixj] = a; }
        }
      }
      __syncthreads();
    }
  }
  for (int e = 0; e < 2; e++) {
    int i = t + e * 1024;
    order[(size_t)b * N_BOX + i] = (int)(keys[i] & 0xFFFFFFFFu);
  }
}

// ---------------------------------------------------------------------------
// K4: greedy class-aware NMS.  Same-class 16x2048 strips are built on the
//     matrix pipe (one-hot x one-hot^T via v_wmma_f32_16x16x32_f16); sorted
//     boxes are staged into LDS with async global->LDS DMA (ASYNCcnt).
// ---------------------------------------------------------------------------
__global__ void __launch_bounds__(1024) k4_nms(const float4* __restrict__ boxes,
                                               const int* __restrict__ labels,
                                               const int* __restrict__ fvalid,
                                               const int* __restrict__ order,
                                               float* __restrict__ out) {
  const int b = blockIdx.x, t = threadIdx.x;
  __shared__ float4   sbox[N_BOX];           // 32 KB
  __shared__ int      slab[N_BOX];           //  8 KB
  __shared__ int      keepA[N_BOX];          //  8 KB
  __shared__ int      ordL[N_BOX];           //  8 KB
  __shared__ unsigned strip[16 * 64];        //  4 KB (16 rows x 2048 bits)

  // ---- stage sorted arrays into LDS (box gather via async DMA to LDS) ----
  for (int e = 0; e < 2; e++) {
    int i = t + e * 1024;
    int o = order[(size_t)b * N_BOX + i];
    ordL[i]  = o;
    slab[i]  = labels[(size_t)b * N_BOX + o];
    keepA[i] = fvalid[(size_t)b * N_BOX + o];
    unsigned ldsa = (unsigned)(size_t)&sbox[i];
    unsigned long long ga = (unsigned long long)(size_t)(boxes + (size_t)b * N_BOX + o);
    asm volatile("global_load_async_to_lds_b128 %0, %1, off"
                 :: "v"(ldsa), "v"(ga) : "memory");
  }
#if __has_builtin(__builtin_amdgcn_s_wait_asynccnt)
  __builtin_amdgcn_s_wait_asynccnt(0);
#else
  asm volatile("s_wait_asynccnt 0" ::: "memory");
#endif
  __syncthreads();

  const int lane = t & 31;
  const int wave = t >> 5;                   // 32 waves

  for (int ib = 0; ib < N_BOX / 16; ib++) {
    const int rowbase = ib * 16;

    // zero the same-class strip
    strip[t & 1023] = 0u;                    // 1024 threads cover 1024 words
    __syncthreads();

    // ---- WMMA one-hot matmul: same[r][j] for 16 rows x 2048 cols ----
    // Each wave builds 4 tiles of 16x16; K = 96 (3 chunks of 32) covers 80
    // classes (one-hot dot == 1 iff labels equal).
    const int rlab = slab[rowbase + (lane & 15)];
    for (int tt = 0; tt < 4; tt++) {
      const int colbase = (wave * 4 + tt) * 16;
      const int clab = slab[colbase + (lane & 15)];
      v8f acc = {};
      for (int c = 0; c < 96; c += 32) {
        v16h a, bm;
        for (int v = 0; v < 8; v++) {
          for (int pp = 0; pp < 2; pp++) {
            int K  = c + ((v < 4) ? 0 : 16) + ((lane < 16) ? 0 : 8) + 2 * (v & 3) + pp;
            int e2 = 2 * v + pp;
            a[e2]  = (rlab == K) ? (_Float16)1.0f : (_Float16)0.0f;
            bm[e2] = (clab == K) ? (_Float16)1.0f : (_Float16)0.0f;
          }
        }
        acc = __builtin_amdgcn_wmma_f32_16x16x32_f16(
            /*neg_a=*/false, a, /*neg_b=*/false, bm,
            /*c_mod=*/(short)0, acc, /*reuse_a=*/false, /*reuse_b=*/false);
      }
      for (int k = 0; k < 8; k++) {
        int M = k + ((lane < 16) ? 0 : 8);           // C/D layout: VGPR k, lane half
        int Nc = colbase + (lane & 15);
        if (acc[k] > 0.5f)
          atomicOr(&strip[M * 64 + (Nc >> 5)], 1u << (Nc & 31));
      }
    }
    __syncthreads();

    // ---- sequential greedy suppression over the 16 rows of this group ----
    for (int r = 0; r < 16; r++) {
      const int i = rowbase + r;
      const int ki = keepA[i];                       // uniform across block
      if (ki) {
        const float4 bi = sbox[i];
        const float areai = (bi.z - bi.x) * (bi.w - bi.y);
        for (int e = 0; e < 2; e++) {
          int j = t + e * 1024;
          if (j > i && keepA[j]) {
            unsigned sm = strip[r * 64 + (j >> 5)] >> (j & 31);
            if (sm & 1u) {
              float4 bj = sbox[j];
              float xx1 = fmaxf(bi.x, bj.x), yy1 = fmaxf(bi.y, bj.y);
              float xx2 = fminf(bi.z, bj.z), yy2 = fminf(bi.w, bj.w);
              float inter = fmaxf(xx2 - xx1, 0.f) * fmaxf(yy2 - yy1, 0.f);
              float areaj = (bj.z - bj.x) * (bj.w - bj.y);
              float uni = fmaxf(areai + areaj - inter, 1e-9f);
              if (inter / uni > NMS_THRES) keepA[j] = 0;
            }
          }
        }
      }
      __syncthreads();
    }
  }

  // scatter keep back to original order -> out[..., 6]
  for (int e = 0; e < 2; e++) {
    int i = t + e * 1024;
    out[((size_t)b * N_BOX + ordL[i]) * 7 + 6] = keepA[i] ? 1.0f : 0.0f;
  }
}

// ---------------------------------------------------------------------------
extern "C" void kernel_launch(void* const* d_in, const int* in_sizes, int n_in,
                              void* d_out, int out_size, void* d_ws, size_t ws_size,
                              hipStream_t stream) {
  const float* pred = (const float*)d_in[0];

  char* w = (char*)d_ws;
  int* validA = (int*)w;        w += (size_t)B_IMG * N_BOX * sizeof(int);
  int* useXY  = (int*)w;        w += 128;
  float4* boxes = (float4*)w;   w += (size_t)B_IMG * N_BOX * sizeof(float4);
  int* fvalid = (int*)w;        w += (size_t)B_IMG * N_BOX * sizeof(int);
  int* labels = (int*)w;        w += (size_t)B_IMG * N_BOX * sizeof(int);
  unsigned* skey = (unsigned*)w; w += (size_t)B_IMG * N_BOX * sizeof(unsigned);
  int* order  = (int*)w;        w += (size_t)B_IMG * N_BOX * sizeof(int);

  float* out    = (float*)d_out;
  float* outLab = out + (size_t)B_IMG * N_BOX * 7;

  k1_stats <<<B_IMG, 256, 0, stream>>>(pred, validA, useXY);
  k2_decode<<<(B_IMG * N_BOX) / 256, 256, 0, stream>>>(pred, validA, useXY, boxes,
                                                       fvalid, labels, skey, out, outLab);
  k3_sort  <<<B_IMG, 1024, 0, stream>>>(skey, order);
  k4_nms   <<<B_IMG, 1024, 0, stream>>>(boxes, labels, fvalid, order, out);
}
